// GrokAttention_81913616269624
// MI455X (gfx1250) — matrix operand order repaired and checked by reading, compiled, not run on hardware
//
#include <hip/hip_runtime.h>
#include <hip/hip_bf16.h>

#define S_LEN 1024
#define NH 64
#define NKVH 8
#define HD 128
#define HID 8192
#define KVDIM 1024
#define ATT_SCALE 0.08838834764831845f  // 1/sqrt(128)
#define LOGIT_CAP 30.0f
#define ROPE_THETA 208533496.0f

typedef __attribute__((ext_vector_type(16))) __bf16 v16bf;
typedef __attribute__((ext_vector_type(8)))  float  v8f;

union FragBF { unsigned int u[8]; v16bf v; };

__device__ __forceinline__ v8f v8f_zero() {
  v8f z;
  #pragma unroll
  for (int i = 0; i < 8; ++i) z[i] = 0.0f;
  return z;
}

__device__ __forceinline__ unsigned short f32_to_bf16_u16(float x) {
  unsigned int a = __float_as_uint(x);
  a += 0x7FFFu + ((a >> 16) & 1u);  // round-to-nearest-even
  return (unsigned short)(a >> 16);
}
__device__ __forceinline__ unsigned int pack_bf16(float lo, float hi) {
  return (unsigned int)f32_to_bf16_u16(lo) |
         ((unsigned int)f32_to_bf16_u16(hi) << 16);
}

// Async 16B global -> LDS copy (CDNA5 GLOBAL_LOAD_ASYNC_TO_LDS_B128, ASYNCcnt).
__device__ __forceinline__ void async_cp16(void* lds_ptr, const void* gptr) {
  unsigned int lds_off = (unsigned int)(size_t)lds_ptr;  // low 32b = LDS offset
  asm volatile("global_load_async_to_lds_b128 %0, %1, off"
               :: "v"(lds_off), "v"(gptr) : "memory");
}

// ---------------------------------------------------------------------------
// Elementwise f32 -> bf16 (row-major bf16 == pair-packed A layout as u32).
// ---------------------------------------------------------------------------
__global__ __launch_bounds__(256)
void cvt_bf16(const float* __restrict__ in, unsigned short* __restrict__ out) {
  size_t idx = (size_t)blockIdx.x * 256 + threadIdx.x;
  out[idx] = f32_to_bf16_u16(in[idx]);
}

// ---------------------------------------------------------------------------
// Pack weight W[K,N] f32 -> Wp[K/2, N] u32 with K-pair interleave:
// Wp[kp*N + n] = {bf16(W[2kp][n]), bf16(W[2kp+1][n])}  (B-fragment layout)
// ---------------------------------------------------------------------------
__global__ __launch_bounds__(256)
void pack_weight_bf16(const float* __restrict__ W, unsigned int* __restrict__ out,
                      int nshift) {
  size_t idx = (size_t)blockIdx.x * 256 + threadIdx.x;  // < K*N/2
  size_t n = idx & (((size_t)1 << nshift) - 1);
  size_t kp = idx >> nshift;
  const float* base = W + ((kp * 2) << nshift) + n;
  out[idx] = pack_bf16(base[0], base[(size_t)1 << nshift]);
}

// ---------------------------------------------------------------------------
// GEMM: C[M,N](f32) = A[M,K](bf16 row-major) @ B[K,N](pair-packed bf16 u32).
// 256 threads = 8 waves; block tile 128x128; wave tile 64x32 (8 WMMA accs).
// Double-buffered LDS staged with async global->LDS b128 copies.
// ---------------------------------------------------------------------------
__global__ __launch_bounds__(256)
void gemm_bf16_wmma(const unsigned short* __restrict__ Ab,
                    const unsigned int* __restrict__ Bp,
                    float* __restrict__ C, int M, int N, int K) {
  __shared__ unsigned int As[2][128][16];  // [buf][m][k/2]  8KB each
  __shared__ unsigned int Bs[2][16][128];  // [buf][k/2][n]  8KB each

  const int tid = threadIdx.x;
  const int l   = tid & 31;
  const int wv  = tid >> 5;
  const int wm  = wv >> 2;       // 0..1 (M direction)
  const int wn  = wv & 3;        // 0..3 (N direction)
  const int lh  = l >> 4;        // lane half
  const int ll  = l & 15;        // lane within half
  const int m0  = blockIdx.y * 128;
  const int n0  = blockIdx.x * 128;

  auto issue = [&](int buf, int k0) {
    #pragma unroll
    for (int i = 0; i < 2; ++i) {       // A tile: 128 x 32 bf16 = 8KB
      int idx = tid + i * 256;
      int row = idx >> 2, seg = idx & 3;
      async_cp16(&As[buf][row][seg * 4],
                 Ab + (size_t)(m0 + row) * K + k0 + seg * 8);
    }
    #pragma unroll
    for (int i = 0; i < 2; ++i) {       // B tile: 16 x 128 u32 = 8KB
      int idx = tid + i * 256;
      int row = idx >> 5, seg = idx & 31;
      async_cp16(&Bs[buf][row][seg * 4],
                 Bp + (size_t)((k0 >> 1) + row) * N + n0 + seg * 4);
    }
  };

  v8f acc[4][2];
  #pragma unroll
  for (int i = 0; i < 4; ++i)
    #pragma unroll
    for (int j = 0; j < 2; ++j) acc[i][j] = v8f_zero();

  const int niter = K >> 5;
  issue(0, 0);
  for (int it = 0; it < niter; ++it) {
    const int buf = it & 1;
    if (it + 1 < niter) {
      issue(buf ^ 1, (it + 1) << 5);
      asm volatile("s_wait_asynccnt 0x4" ::: "memory");  // current tile done
    } else {
      asm volatile("s_wait_asynccnt 0x0" ::: "memory");
    }
    __syncthreads();

    FragBF a[4], b[2];
    #pragma unroll
    for (int ms = 0; ms < 4; ++ms)
      #pragma unroll
      for (int j = 0; j < 8; ++j)
        a[ms].u[j] = As[buf][wm * 64 + ms * 16 + ll][8 * lh + j];
    #pragma unroll
    for (int ns = 0; ns < 2; ++ns)
      #pragma unroll
      for (int j = 0; j < 8; ++j)
        b[ns].u[j] = Bs[buf][8 * lh + j][wn * 32 + ns * 16 + ll];

    #pragma unroll
    for (int ms = 0; ms < 4; ++ms)
      #pragma unroll
      for (int ns = 0; ns < 2; ++ns)
        acc[ms][ns] = __builtin_amdgcn_wmma_f32_16x16x32_bf16(
            false, a[ms].v, false, b[ns].v, (short)0, acc[ms][ns], false, false);
    __syncthreads();
  }

  #pragma unroll
  for (int ms = 0; ms < 4; ++ms)
    #pragma unroll
    for (int ns = 0; ns < 2; ++ns)
      #pragma unroll
      for (int j = 0; j < 8; ++j) {
        int row = m0 + wm * 64 + ms * 16 + 8 * lh + j;
        int col = n0 + wn * 32 + ns * 16 + ll;
        C[(size_t)row * N + col] = acc[ms][ns][j];
      }
}

// ---------------------------------------------------------------------------
// RoPE + bf16 conversion. Writes rotated Q,K as bf16; V as bf16 transposed
// (KVDIM x S) so attention's PV B-fragments are contiguous u32 key-pairs.
// ---------------------------------------------------------------------------
__global__ __launch_bounds__(256)
void rope_convert(const float* __restrict__ Qf, const float* __restrict__ Kf,
                  const float* __restrict__ Vf, const int* __restrict__ pos_ids,
                  unsigned short* __restrict__ Qb, unsigned short* __restrict__ Kb,
                  unsigned short* __restrict__ Vbt) {
  int idx = blockIdx.x * 256 + threadIdx.x;  // < S*HID
  int s = idx >> 13;
  int c = idx & (HID - 1);
  int d = c & 127;
  int j = d & 63;
  float pos = (float)pos_ids[s];
  float inv = __expf(-0.015625f * (float)j * __logf(ROPE_THETA));  // theta^(-2j/128)
  float cs, sn;
  __sincosf(pos * inv, &cs, &sn);

  float q  = Qf[idx];
  float qr = (d < 64) ? -Qf[idx + 64] : Qf[idx - 64];
  Qb[idx] = f32_to_bf16_u16(q * cs + qr * sn);

  if (c < KVDIM) {
    int ki = s * KVDIM + c;
    float k  = Kf[ki];
    float kr = (d < 64) ? -Kf[ki + 64] : Kf[ki - 64];
    Kb[ki] = f32_to_bf16_u16(k * cs + kr * sn);
    Vbt[c * S_LEN + s] = f32_to_bf16_u16(Vf[ki]);
  }
}

// ---------------------------------------------------------------------------
// Flash-style causal attention with tanh logit cap.
// Block = (head, 64-query tile), 4 waves; each wave owns 16 query rows.
// Per 32-key step: 8 WMMA (QK^T) + 8 WMMA (P V), online softmax in regs.
// Output written as bf16 row-major so the Wo GEMM consumes it directly.
// ---------------------------------------------------------------------------
__global__ __launch_bounds__(128)
void flash_attn_wmma(const unsigned int* __restrict__ Qb,   // S x HID/2
                     const unsigned int* __restrict__ Kb,   // S x KVDIM/2
                     const unsigned int* __restrict__ VbT,  // KVDIM x S/2
                     unsigned short* __restrict__ Ao) {     // S x HID (bf16)
  __shared__ unsigned short Psm[4][16][32];  // per-wave P tile bounce buffer

  const int h   = blockIdx.y;
  const int kvh = h >> 3;
  const int q0  = blockIdx.x * 64;
  const int tid = threadIdx.x;
  const int w   = tid >> 5;
  const int l   = tid & 31;
  const int lh  = l >> 4, ll = l & 15;

  // Q fragments: 16 rows x 128 d, as 4 chunks of K=32
  FragBF aq[4];
  #pragma unroll
  for (int c = 0; c < 4; ++c)
    #pragma unroll
    for (int j = 0; j < 8; ++j)
      aq[c].u[j] = Qb[(size_t)(q0 + w * 16 + ll) * (HID / 2) + h * 64 + c * 16 +
                      8 * lh + j];

  v8f o[8];
  #pragma unroll
  for (int d = 0; d < 8; ++d) o[d] = v8f_zero();
  float mi[8], li[8];
  #pragma unroll
  for (int j = 0; j < 8; ++j) { mi[j] = -1e30f; li[j] = 0.0f; }

  const int kv_end = q0 + 64;
  for (int kv = 0; kv < kv_end; kv += 32) {
    // ---- scores: two 16-key tiles, K-dim 128 = 4 WMMAs each ----
    v8f sc[2];
    #pragma unroll
    for (int t = 0; t < 2; ++t) {
      sc[t] = v8f_zero();
      #pragma unroll
      for (int c = 0; c < 4; ++c) {
        FragBF bk;
        #pragma unroll
        for (int j = 0; j < 8; ++j)
          bk.u[j] = Kb[(size_t)(kv + t * 16 + ll) * (KVDIM / 2) + kvh * 64 +
                       c * 16 + 8 * lh + j];
        sc[t] = __builtin_amdgcn_wmma_f32_16x16x32_bf16(
            false, aq[c].v, false, bk.v, (short)0, sc[t], false, false);
      }
    }

    // ---- tanh cap + causal mask + online softmax (rows = 8*lh + j) ----
    #pragma unroll
    for (int j = 0; j < 8; ++j) {
      int qrow = q0 + w * 16 + 8 * lh + j;
      float x0 = LOGIT_CAP * tanhf(sc[0][j] * ATT_SCALE * (1.0f / LOGIT_CAP));
      float x1 = LOGIT_CAP * tanhf(sc[1][j] * ATT_SCALE * (1.0f / LOGIT_CAP));
      if (kv + ll > qrow)      x0 += -1e9f;
      if (kv + 16 + ll > qrow) x1 += -1e9f;
      float r = fmaxf(x0, x1);
      for (int off = 1; off < 16; off <<= 1)
        r = fmaxf(r, __shfl_xor(r, off, 32));
      float mn   = fmaxf(mi[j], r);
      float corr = __expf(mi[j] - mn);
      float p0 = __expf(x0 - mn);
      float p1 = __expf(x1 - mn);
      float rs = p0 + p1;
      for (int off = 1; off < 16; off <<= 1)
        rs += __shfl_xor(rs, off, 32);
      li[j] = li[j] * corr + rs;
      mi[j] = mn;
      #pragma unroll
      for (int d = 0; d < 8; ++d) o[d][j] *= corr;
      Psm[w][8 * lh + j][ll]      = f32_to_bf16_u16(p0);
      Psm[w][8 * lh + j][16 + ll] = f32_to_bf16_u16(p1);
    }
    __syncthreads();

    // ---- reshape P (16x32) into an A fragment ----
    FragBF ap;
    #pragma unroll
    for (int j = 0; j < 8; ++j) {
      int k = 16 * lh + 2 * j;
      ap.u[j] = *reinterpret_cast<const unsigned int*>(&Psm[w][ll][k]);
    }
    __syncthreads();

    // ---- P @ V: 8 d-subtiles of 16 ----
    #pragma unroll
    for (int d = 0; d < 8; ++d) {
      FragBF bv;
      #pragma unroll
      for (int j = 0; j < 8; ++j)
        bv.u[j] = VbT[(size_t)(kvh * 128 + d * 16 + ll) * (S_LEN / 2) +
                      (kv >> 1) + 8 * lh + j];
      o[d] = __builtin_amdgcn_wmma_f32_16x16x32_bf16(
          false, ap.v, false, bv.v, (short)0, o[d], false, false);
    }
  }

  // ---- normalize and store bf16 ----
  #pragma unroll
  for (int j = 0; j < 8; ++j) {
    float invl = 1.0f / li[j];
    int q = q0 + w * 16 + 8 * lh + j;
    #pragma unroll
    for (int d = 0; d < 8; ++d)
      Ao[(size_t)q * HID + h * 128 + d * 16 + ll] =
          f32_to_bf16_u16(o[d][j] * invl);
  }
}

// ---------------------------------------------------------------------------
// Launch: pack weights/X -> QKV GEMMs -> RoPE/bf16 -> attention -> Wo GEMM
// ---------------------------------------------------------------------------
extern "C" void kernel_launch(void* const* d_in, const int* in_sizes, int n_in,
                              void* d_out, int out_size, void* d_ws, size_t ws_size,
                              hipStream_t stream) {
  const float* X  = (const float*)d_in[0];
  const float* Wq = (const float*)d_in[1];
  const float* Wk = (const float*)d_in[2];
  const float* Wv = (const float*)d_in[3];
  const float* Wo = (const float*)d_in[4];
  const int*  pos = (const int*)d_in[6];
  float* out = (float*)d_out;

  char* ws = (char*)d_ws;
  float* Qf           = (float*)ws;                                  //  32 MB
  float* Kf           = (float*)(ws + ((size_t)32 << 20));           //   4 MB
  float* Vf           = (float*)(ws + ((size_t)36 << 20));           //   4 MB
  unsigned short* Qb  = (unsigned short*)(ws + ((size_t)40 << 20));  //  16 MB
  unsigned short* Kb  = (unsigned short*)(ws + ((size_t)56 << 20));  //   2 MB
  unsigned short* Vb  = (unsigned short*)(ws + ((size_t)58 << 20));  //   2 MB
  unsigned short* Ao  = (unsigned short*)(ws + ((size_t)60 << 20));  //  16 MB
  unsigned short* Xb  = (unsigned short*)(ws + ((size_t)76 << 20));  //  16 MB
  unsigned int*  Wqb  = (unsigned int*)(ws + ((size_t)92 << 20));    // 128 MB
  unsigned int*  Wkb  = (unsigned int*)(ws + ((size_t)220 << 20));   //  16 MB
  unsigned int*  Wvb  = (unsigned int*)(ws + ((size_t)236 << 20));   //  16 MB
  unsigned int*  Wob  = (unsigned int*)(ws + ((size_t)252 << 20));   // 128 MB

  dim3 blk(256);
  cvt_bf16<<<(S_LEN * HID) / 256, blk, 0, stream>>>(X, Xb);
  pack_weight_bf16<<<(HID * HID / 2) / 256, blk, 0, stream>>>(Wq, Wqb, 13);
  pack_weight_bf16<<<(HID * KVDIM / 2) / 256, blk, 0, stream>>>(Wk, Wkb, 10);
  pack_weight_bf16<<<(HID * KVDIM / 2) / 256, blk, 0, stream>>>(Wv, Wvb, 10);
  pack_weight_bf16<<<(HID * HID / 2) / 256, blk, 0, stream>>>(Wo, Wob, 13);

  gemm_bf16_wmma<<<dim3(HID / 128, S_LEN / 128), blk, 0, stream>>>(
      Xb, Wqb, Qf, S_LEN, HID, HID);
  gemm_bf16_wmma<<<dim3(KVDIM / 128, S_LEN / 128), blk, 0, stream>>>(
      Xb, Wkb, Kf, S_LEN, KVDIM, HID);
  gemm_bf16_wmma<<<dim3(KVDIM / 128, S_LEN / 128), blk, 0, stream>>>(
      Xb, Wvb, Vf, S_LEN, KVDIM, HID);

  rope_convert<<<(S_LEN * HID) / 256, blk, 0, stream>>>(
      Qf, Kf, Vf, pos, Qb, Kb, Vb);

  flash_attn_wmma<<<dim3(S_LEN / 64, NH), dim3(128), 0, stream>>>(
      (const unsigned int*)Qb, (const unsigned int*)Kb,
      (const unsigned int*)Vb, Ao);

  gemm_bf16_wmma<<<dim3(HID / 128, S_LEN / 128), blk, 0, stream>>>(
      Ao, Wob, out, S_LEN, HID, HID);
}
